// Encoder_64707977282277
// MI455X (gfx1250) — matrix operand order repaired and checked by reading, not compiled, not run on hardware
//
#include <hip/hip_runtime.h>

typedef __attribute__((ext_vector_type(16))) __bf16 v16bf;
typedef __attribute__((ext_vector_type(8)))  __bf16 v8bf;
typedef __attribute__((ext_vector_type(4)))  __bf16 v4bf;
typedef __attribute__((ext_vector_type(8)))  float  v8f;

#define HID 64
#define BATCH 8
#define TSTEPS 16
#define HW 32
#define PIX_PER_IMG (HW*HW)            // 1024
#define PIX_TOTAL (BATCH*PIX_PER_IMG)  // 8192
#define KHH (49*64)                    // 3136: recurrent-conv K
#define KIH 224                        // input-conv K (56 padded taps * 4 padded ch)
#define LDS_CS 72                      // padded channel stride in h-halo LDS
#define HALO_W 38
#define HALO_H 8

static __device__ __forceinline__ unsigned short f2bf(float f) {
  union { float f; unsigned u; } v; v.f = f;
  return (unsigned short)((v.u + 0x7FFFu + ((v.u >> 16) & 1u)) >> 16);
}
static __device__ __forceinline__ float sigm(float x) { return 1.f / (1.f + __expf(-x)); }
static __device__ __forceinline__ float tanh_fast(float x) { return 1.f - 2.f / (__expf(2.f * x) + 1.f); }

// ---- weight transform: w_hh [256][64][7][7] f32 -> bf16 [dir][256][tap*64+cin] ----
__global__ void k_prep_whh(const float* __restrict__ wf, const float* __restrict__ wb,
                           unsigned short* __restrict__ dst) {
  int i = blockIdx.x * blockDim.x + threadIdx.x;
  const int per = 256 * KHH;
  if (i >= 2 * per) return;
  int d = i / per, e = i - d * per;
  int n = e / KHH, k = e - n * KHH;
  int tap = k >> 6, cin = k & 63;
  const float* src = d ? wb : wf;
  dst[i] = f2bf(src[(size_t)n * (64 * 49) + cin * 49 + tap]);
}

// ---- weight transform: w_ih [256][3][7][7] f32 -> bf16 [dir][256][tap*4+ci], zero padded ----
__global__ void k_prep_wih(const float* __restrict__ wf, const float* __restrict__ wb,
                           unsigned short* __restrict__ dst) {
  int i = blockIdx.x * blockDim.x + threadIdx.x;
  const int per = 256 * KIH;
  if (i >= 2 * per) return;
  int d = i / per, e = i - d * per;
  int n = e / KIH, k = e - n * KIH;
  int tap = k >> 2, ci = k & 3;
  const float* src = d ? wb : wf;
  float v = (ci < 3 && tap < 49) ? src[(size_t)n * 147 + ci * 49 + tap] : 0.f;
  dst[i] = f2bf(v);
}

// ---- x [B][T][3][32][32] f32 -> bf16 NHWC4 [B*T][1024][4] ----
__global__ void k_prep_x(const float* __restrict__ x, unsigned short* __restrict__ dst) {
  int i = blockIdx.x * blockDim.x + threadIdx.x;
  if (i >= 131072 * 4) return;
  int p = i >> 2, ci = i & 3;
  int bt = p >> 10, sp = p & 1023;
  float v = (ci < 3) ? x[(size_t)bt * 3072 + ci * 1024 + sp] : 0.f;
  dst[i] = f2bf(v);
}

// ---- fused per-timestep kernel: implicit-GEMM conv (WMMA bf16) + LSTM gating ----
__global__ __launch_bounds__(256) void k_step(
    int s,
    const unsigned short* __restrict__ whh_t, const unsigned short* __restrict__ wih_t,
    const unsigned short* __restrict__ x4,
    unsigned short* __restrict__ hbuf, float* __restrict__ cbuf,
    const float* __restrict__ bihf, const float* __restrict__ bhhf,
    const float* __restrict__ bihb, const float* __restrict__ bhhb,
    float* __restrict__ out) {
  __shared__ __bf16 lds_h[HALO_H * HALO_W * LDS_CS];  // 43776 B
  __shared__ __bf16 lds_x[HALO_H * HALO_W * 4];       //  2432 B
  __shared__ float  lds_t[64 * 65];                   // 16640 B

  const int tid = threadIdx.x;
  const int lane = tid & 31;
  const int wave = tid >> 5;
  const int wm = wave >> 2;   // m-half: pixels [wm*32, wm*32+32)
  const int wc = wave & 3;    // c-tile: hidden channels [wc*16, wc*16+16)
  const int l15 = lane & 15;
  const int lh = lane >> 4;

  const int dir = blockIdx.y;
  const int mb = blockIdx.x;            // 0..127
  const int bidx = mb >> 4;             // batch
  const int y0 = (mb & 15) << 1;        // two image rows per block
  const int t = dir ? (TSTEPS - 1 - s) : s;

  const __bf16* whh = (const __bf16*)(whh_t + (size_t)dir * 256 * KHH);
  const __bf16* wih = (const __bf16*)(wih_t + (size_t)dir * 256 * KIH);
  const unsigned short* hprev = hbuf + (size_t)(dir * 2 + (s & 1)) * (PIX_TOTAL * HID);
  unsigned short* hnext = hbuf + (size_t)(dir * 2 + ((s + 1) & 1)) * (PIX_TOTAL * HID);
  float* cw = cbuf + (size_t)dir * PIX_TOTAL * HID;
  const float* bih = dir ? bihb : bihf;
  const float* bhh = dir ? bhhb : bhhf;

  // stage h_prev halo (bf16 NHWC, zero-padded; zeros at s==0)
  for (int i = tid; i < HALO_H * HALO_W * 16; i += 256) {
    int ry = i / (HALO_W * 16);
    int r = i - ry * (HALO_W * 16);
    int cx = r >> 4;
    int cq = r & 15;
    int y = y0 - 3 + ry, xx = cx - 3;
    union { v4bf v; unsigned long long u; } q; q.u = 0ull;
    if (s > 0 && y >= 0 && y < HW && xx >= 0 && xx < HW)
      q.v = *(const v4bf*)&hprev[((size_t)(bidx * PIX_PER_IMG + y * HW + xx)) * HID + cq * 4];
    *(v4bf*)&lds_h[(ry * HALO_W + cx) * LDS_CS + cq * 4] = q.v;
  }
  // stage x halo (4 padded channels)
  for (int i = tid; i < HALO_H * HALO_W; i += 256) {
    int ry = i / HALO_W, cx = i - ry * HALO_W;
    int y = y0 - 3 + ry, xx = cx - 3;
    union { v4bf v; unsigned long long u; } q; q.u = 0ull;
    if (y >= 0 && y < HW && xx >= 0 && xx < HW)
      q.v = *(const v4bf*)&x4[((size_t)((bidx * TSTEPS + t) * PIX_PER_IMG + y * HW + xx)) * 4];
    *(v4bf*)&lds_x[i * 4] = q.v;
  }
  __syncthreads();

  const v8f zero8 = {0.f, 0.f, 0.f, 0.f, 0.f, 0.f, 0.f, 0.f};
  v8f acc[2][4];
#pragma unroll
  for (int mt = 0; mt < 2; ++mt)
#pragma unroll
    for (int g = 0; g < 4; ++g) acc[mt][g] = zero8;

  const int m0 = wm * 32 + l15;

  // ---- recurrent conv: 49 taps x (2 halves of 32 channels) = 98 K-steps ----
  for (int tap = 0; tap < 49; ++tap) {
    const int dyp = tap / 7, dxp = tap % 7;
#pragma unroll
    for (int half = 0; half < 2; ++half) {
      const int ks = tap * 2 + half;
      v16bf afr[2];
#pragma unroll
      for (int mt = 0; mt < 2; ++mt) {
        int m = m0 + mt * 16;
        int base = (((m >> 5) + dyp) * HALO_W + (m & 31) + dxp) * LDS_CS + half * 32 + lh * 16;
        union { v16bf v; v8bf h[2]; } A;
        A.h[0] = *(const v8bf*)&lds_h[base];        // K[lh*16 .. +7]
        A.h[1] = *(const v8bf*)&lds_h[base + 8];    // K[lh*16+8 .. +7]
        afr[mt] = A.v;
      }
      v16bf bfr[4];
#pragma unroll
      for (int g = 0; g < 4; ++g)
        bfr[g] = *(const v16bf*)&whh[(size_t)(wc * 16 + g * 64 + l15) * KHH + ks * 32 + lh * 16];
#pragma unroll
      for (int mt = 0; mt < 2; ++mt)
#pragma unroll
        for (int g = 0; g < 4; ++g)
          acc[mt][g] = __builtin_amdgcn_wmma_f32_16x16x32_bf16(
              false, afr[mt], false, bfr[g], (short)0, acc[mt][g], false, false);
    }
  }

  // ---- input conv fused in: 7 K-steps over 56 padded taps x 4 padded channels ----
  for (int ksx = 0; ksx < 7; ++ksx) {
    v16bf afr[2];
#pragma unroll
    for (int mt = 0; mt < 2; ++mt) {
      int m = m0 + mt * 16;
      union { v16bf v; v4bf q[4]; } A;
#pragma unroll
      for (int j = 0; j < 4; ++j) {
        int tl = lh * 4 + j;   // tap-in-tile per A-layout chunks
        int tap = ksx * 8 + tl;
        union { v4bf v; unsigned long long u; } q; q.u = 0ull;
        if (tap < 49) {
          int dyp = tap / 7, dxp = tap % 7;
          q.v = *(const v4bf*)&lds_x[(((m >> 5) + dyp) * HALO_W + (m & 31) + dxp) * 4];
        }
        A.q[j] = q.v;
      }
      afr[mt] = A.v;
    }
    v16bf bfr[4];
#pragma unroll
    for (int g = 0; g < 4; ++g)
      bfr[g] = *(const v16bf*)&wih[(size_t)(wc * 16 + g * 64 + l15) * KIH + ksx * 32 + lh * 16];
#pragma unroll
    for (int mt = 0; mt < 2; ++mt)
#pragma unroll
      for (int g = 0; g < 4; ++g)
        acc[mt][g] = __builtin_amdgcn_wmma_f32_16x16x32_bf16(
            false, afr[mt], false, bfr[g], (short)0, acc[mt][g], false, false);
  }

  // ---- LSTM gating: each lane holds all 4 gates for (pixel, channel) ----
  const int cc = wc * 16 + l15;
  const float bias0 = bih[cc] + bhh[cc];
  const float bias1 = bih[64 + cc] + bhh[64 + cc];
  const float bias2 = bih[128 + cc] + bhh[128 + cc];
  const float bias3 = bih[192 + cc] + bhh[192 + cc];

#pragma unroll
  for (int mt = 0; mt < 2; ++mt) {
#pragma unroll
    for (int j = 0; j < 8; ++j) {
      int m = wm * 32 + mt * 16 + lh * 8 + j;       // C-layout row
      size_t p = (size_t)bidx * PIX_PER_IMG + y0 * HW + m;
      float iv = sigm(acc[mt][0][j] + bias0);
      float fv = sigm(acc[mt][1][j] + bias1);
      float gv = tanh_fast(acc[mt][2][j] + bias2);
      float ov = sigm(acc[mt][3][j] + bias3);
      float cp = (s == 0) ? 0.f : cw[p * HID + cc];
      float cn = fv * cp + iv * gv;
      cw[p * HID + cc] = cn;
      float hn = ov * tanh_fast(cn);
      hnext[p * HID + cc] = f2bf(hn);
      lds_t[m * 65 + cc] = hn;
    }
  }
  __syncthreads();

  // coalesced NCHW f32 output store via LDS transpose
  for (int it = 0; it < 16; ++it) {
    int idx = it * 256 + tid;
    int ch = idx >> 6;
    int m = idx & 63;
    out[((size_t)((bidx * TSTEPS + t) * 128 + dir * 64 + ch)) * PIX_PER_IMG + y0 * HW + m] =
        lds_t[m * 65 + ch];
  }
}

extern "C" void kernel_launch(void* const* d_in, const int* in_sizes, int n_in,
                              void* d_out, int out_size, void* d_ws, size_t ws_size,
                              hipStream_t stream) {
  const float* x    = (const float*)d_in[0];
  const float* wihf = (const float*)d_in[1];
  const float* whhf = (const float*)d_in[2];
  const float* bihf = (const float*)d_in[3];
  const float* bhhf = (const float*)d_in[4];
  const float* wihb = (const float*)d_in[5];
  const float* whhb = (const float*)d_in[6];
  const float* bihb = (const float*)d_in[7];
  const float* bhhb = (const float*)d_in[8];
  float* out = (float*)d_out;

  unsigned short* ws = (unsigned short*)d_ws;
  size_t o = 0;
  unsigned short* whh_t = ws + o; o += (size_t)2 * 256 * KHH;        // 3.2 MB
  unsigned short* wih_t = ws + o; o += (size_t)2 * 256 * KIH;        // 229 KB
  unsigned short* x4    = ws + o; o += (size_t)131072 * 4;           // 1 MB
  unsigned short* hbuf  = ws + o; o += (size_t)2 * 2 * PIX_TOTAL * HID; // 4 MB (ping-pong x 2 dirs)
  float* cbuf = (float*)(ws + o);                                    // 4 MB

  const int thr = 256;
  k_prep_whh<<<(2 * 256 * KHH + thr - 1) / thr, thr, 0, stream>>>(whhf, whhb, whh_t);
  k_prep_wih<<<(2 * 256 * KIH + thr - 1) / thr, thr, 0, stream>>>(wihf, wihb, wih_t);
  k_prep_x<<<(131072 * 4 + thr - 1) / thr, thr, 0, stream>>>(x, x4);

  for (int s = 0; s < TSTEPS; ++s) {
    k_step<<<dim3(128, 2), 256, 0, stream>>>(s, whh_t, wih_t, x4, hbuf, cbuf,
                                             bihf, bhhf, bihb, bhhb, out);
  }
}
